// PointTransformer_31963146616903
// MI455X (gfx1250) — matrix operand order repaired and checked by reading, not compiled, not run on hardware
//
#include <hip/hip_runtime.h>
#include <hip/hip_bf16.h>
#include <cstdint>
#include <cstddef>

typedef _Float16 half_t;
typedef __attribute__((ext_vector_type(16))) _Float16 v16h;
typedef __attribute__((ext_vector_type(8)))  _Float16 v8h;
typedef __attribute__((ext_vector_type(4)))  _Float16 v4h;
typedef __attribute__((ext_vector_type(2)))  _Float16 v2h;
typedef __attribute__((ext_vector_type(8)))  float    v8f;
typedef __attribute__((ext_vector_type(4)))  float    v4f;
typedef __attribute__((ext_vector_type(2)))  float    v2f;

#define DEV __device__ __forceinline__

// Padded row stride (halves) for f16 A-matrices in LDS: 72 halves = 144 B.
// Row base bank = (36*m) % 64 -> 16 rows hit 16 disjoint 4-bank groups (conflict-free b128).
#define AST 72
// Swizzled B-weight storage: fragment f = kt*4+nt, lane chunk = 16 halves padded to 24.
#define BCHUNK 24
#define WSW_MAT (8 * 32 * BCHUNK)   // 6144 halves per 64x64 matrix

// Edge-kernel tiling: 8 destination points/block -> 136 edge rows -> 9 M-tiles (144 rows).
#define TP    8
#define NEDG  (TP * 17)     // 136
#define NROW  144           // padded to 9 * 16
#define NTIL  9

// ---------------- WMMA fragment helpers (gfx1250, wave32) ----------------
// 16-bit A-matrix 16x32 (ISA 7.12.2): lanes 0-15 -> M=lane, K=0..7 (elems 0..7),
// K=16..23 (elems 8..15); lanes 16-31 -> M=lane-16, K=8..15 and 24..31.
DEV v16h load_a_frag(const half_t* __restrict__ buf, int stride, int row0, int k0, int lane) {
  const int m  = row0 + (lane & 15);
  const int kb = k0 + ((lane >> 4) << 3);
  const half_t* p = buf + (size_t)m * stride + kb;   // 16B-aligned
  v8h lo = *(const v8h*)p;
  v8h hi = *(const v8h*)(p + 16);
  return __builtin_shufflevector(lo, hi, 0, 1, 2, 3, 4, 5, 6, 7,
                                 8, 9, 10, 11, 12, 13, 14, 15);
}

// B fragment from pre-swizzled weights: contiguous 16 halves per lane (24-half chunk stride).
DEV v16h load_b_frag(const half_t* __restrict__ wS, int frag, int lane) {
  const half_t* p = wS + (size_t)(frag * 32 + lane) * BCHUNK;  // 48B stride, 16B aligned
  v8h lo = *(const v8h*)p;
  v8h hi = *(const v8h*)(p + 8);
  return __builtin_shufflevector(lo, hi, 0, 1, 2, 3, 4, 5, 6, 7,
                                 8, 9, 10, 11, 12, 13, 14, 15);
}

// D/C f32 16x16: VGPR r -> M = r + 8*(lane>=16), N = lane&15.
DEV void store_tile_f16_lds(half_t* out, int stride, int row0, const v8f acc[4], int lane, bool relu) {
  const int mb = row0 + ((lane >> 4) << 3);
  const int n  = lane & 15;
#pragma unroll
  for (int nt = 0; nt < 4; ++nt)
#pragma unroll
    for (int r = 0; r < 8; ++r) {
      float v = acc[nt][r];
      if (relu) v = v > 0.0f ? v : 0.0f;
      out[(size_t)(mb + r) * stride + nt * 16 + n] = (half_t)v;
    }
}

DEV void store_tile_f32_lds(float* out, int stride, int row0, const v8f acc[4], int lane) {
  const int mb = row0 + ((lane >> 4) << 3);
  const int n  = lane & 15;
#pragma unroll
  for (int nt = 0; nt < 4; ++nt)
#pragma unroll
    for (int r = 0; r < 8; ++r)
      out[(size_t)(mb + r) * stride + nt * 16 + n] = acc[nt][r];
}

DEV void store_tile_f16_global(half_t* g, int row_base, int row0, int nrows, const v8f acc[4], int lane) {
  const int mb = row0 + ((lane >> 4) << 3);
  const int n  = lane & 15;
#pragma unroll
  for (int nt = 0; nt < 4; ++nt)
#pragma unroll
    for (int r = 0; r < 8; ++r) {
      int row = row_base + mb + r;
      if (row < nrows) g[(size_t)row * 64 + nt * 16 + n] = (half_t)acc[nt][r];
    }
}

DEV void store_tile_f32_global(float* g, int row_base, int row0, int nrows, const v8f acc[4], int lane) {
  const int mb = row0 + ((lane >> 4) << 3);
  const int n  = lane & 15;
#pragma unroll
  for (int nt = 0; nt < 4; ++nt)
#pragma unroll
    for (int r = 0; r < 8; ++r) {
      int row = row_base + mb + r;
      if (row < nrows) g[(size_t)row * 64 + nt * 16 + n] = acc[nt][r];
    }
}

// One 16x64 tile: A[16x64] (f16 LDS, stride AST) @ W (pre-swizzled f16 LDS) + bias.
DEV void gemm_tile_64(const half_t* __restrict__ A, int astride, int row0,
                      const half_t* __restrict__ wS, const float* __restrict__ bias,
                      int lane, v8f acc[4]) {
#pragma unroll
  for (int nt = 0; nt < 4; ++nt) {
    float bv = bias ? bias[nt * 16 + (lane & 15)] : 0.0f;
#pragma unroll
    for (int j = 0; j < 8; ++j) acc[nt][j] = bv;
  }
#pragma unroll
  for (int kt = 0; kt < 2; ++kt) {
    v16h a = load_a_frag(A, astride, row0, kt * 32, lane);
#pragma unroll
    for (int nt = 0; nt < 4; ++nt) {
      v16h b = load_b_frag(wS, kt * 4 + nt, lane);
      acc[nt] = __builtin_amdgcn_wmma_f32_16x16x32_f16(
          false, a, false, b, (short)0, acc[nt], false, false);
    }
  }
}

// ---------------- Kernel 0: convert + swizzle 8 weight matrices into B-fragment order ----
__global__ void wcvt_kernel(const float* s0, const float* s1, const float* s2, const float* s3,
                            const float* s4, const float* s5, const float* s6, const float* s7,
                            half_t* dst) {
  const float* srcs[8] = {s0, s1, s2, s3, s4, s5, s6, s7};
  const float* s = srcs[blockIdx.x];
  half_t* d = dst + (size_t)blockIdx.x * WSW_MAT;
  for (int idx = threadIdx.x; idx < 4096; idx += blockDim.x) {
    int j    = idx & 15;
    int lane = (idx >> 4) & 31;
    int nt   = (idx >> 9) & 3;
    int kt   = idx >> 11;
    int k = kt * 32 + ((lane >> 4) << 3) + (j & 7) + ((j >> 3) << 4);
    int n = nt * 16 + (lane & 15);
    d[(size_t)(((kt * 4 + nt) * 32 + lane)) * BCHUNK + j] = (half_t)s[(size_t)k * 64 + n];
  }
}

// ---------------- Kernel 1: h = x@W_in + b_in ; v/a_src/a_dst = h@{W_lin,W_src,W_dst} ----
__global__ void node_kernel(const float* __restrict__ x, const float* __restrict__ b_in,
                            const half_t* __restrict__ w16,
                            half_t* __restrict__ vh, half_t* __restrict__ ash,
                            half_t* __restrict__ adh, int N) {
  extern __shared__ char smem[];
  half_t* xh  = (half_t*)smem;            // 128*AST
  half_t* hh  = xh + 128 * AST;           // 128*AST
  half_t* w4  = hh + 128 * AST;           // 4 * WSW_MAT
  float*  bin = (float*)(w4 + 4 * WSW_MAT);
  const int tid = threadIdx.x, lane = tid & 31, wave = tid >> 5;
  const int rb = blockIdx.x * 128;

  for (int i = tid; i < 4 * WSW_MAT; i += 256) w4[i] = w16[i];
  if (tid < 64) bin[tid] = b_in[tid];
  for (int i = tid; i < 8192; i += 256) {
    int r = i >> 6, c = i & 63;
    int gr = rb + r; if (gr >= N) gr = N - 1;
    xh[(size_t)r * AST + c] = (half_t)x[(size_t)gr * 64 + c];
  }
  __syncthreads();

  v8f acc[4];
  gemm_tile_64(xh, AST, wave * 16, w4, bin, lane, acc);
  store_tile_f16_lds(hh, AST, wave * 16, acc, lane, false);
  __syncthreads();

  gemm_tile_64(hh, AST, wave * 16, w4 + 1 * WSW_MAT, nullptr, lane, acc);
  store_tile_f16_global(vh, rb, wave * 16, N, acc, lane);
  gemm_tile_64(hh, AST, wave * 16, w4 + 2 * WSW_MAT, nullptr, lane, acc);
  store_tile_f16_global(ash, rb, wave * 16, N, acc, lane);
  gemm_tile_64(hh, AST, wave * 16, w4 + 3 * WSW_MAT, nullptr, lane, acc);
  store_tile_f16_global(adh, rb, wave * 16, N, acc, lane);
}

// ---------------- Kernel 2: fused edge pipeline ----------------
// Block = TP=8 destination points = 136 edge rows = 9 WMMA M-tiles (padded to 144 rows).
// ~137 KB LDS -> 2 workgroups per WGP so element phases overlap WMMA phases across blocks.
__global__ void edge_kernel(const float* __restrict__ pos, const int* __restrict__ esrc,
                            const half_t* __restrict__ ash, const half_t* __restrict__ adh,
                            const half_t* __restrict__ vh,
                            const float* __restrict__ posW1, const float* __restrict__ posb1,
                            const float* __restrict__ posb2, const float* __restrict__ attnb1,
                            const float* __restrict__ attnb2,
                            const half_t* __restrict__ w16, half_t* __restrict__ tmph,
                            int N, int K) {
  extern __shared__ char smem[];
  float* bufD  = (float*)smem;            // NROW*64 : delta (f32)
  float* bufL  = bufD + NROW * 64;        // NROW*64 : logits / exp
  float* dAdst = bufL + NROW * 64;        // TP*64
  float* dRel  = dAdst + TP * 64;         // NEDG*3
  float* dPW1  = dRel + NEDG * 3;         // 3*64
  float* dPb1  = dPW1 + 192;              // 64
  float* dPb2  = dPb1 + 64;               // 64
  float* dAb1  = dPb2 + 64;               // 64
  float* dAb2  = dAb1 + 64;               // 64
  int*   srcIdx = (int*)(dAb2 + 64);      // NEDG
  half_t* bufA = (half_t*)(srcIdx + NEDG); // NROW*AST (hidden1 -> e -> hidden2)
  half_t* wB   = bufA + NROW * AST;        // 3 * WSW_MAT (pos_W2, attn_W1, attn_W2)

  const int tid = threadIdx.x, lane = tid & 31, wave = tid >> 5;
  const int p0 = blockIdx.x * TP;
  const int KE = K + 1;                   // 17 (self loop first)

  // ---- loads ----
  for (int i = tid; i < 3 * WSW_MAT; i += 256) wB[i] = w16[4 * WSW_MAT + i];
  if (tid < 192) dPW1[tid] = posW1[tid];
  if (tid < 64) {
    dPb1[tid] = posb1[tid]; dPb2[tid] = posb2[tid];
    dAb1[tid] = attnb1[tid]; dAb2[tid] = attnb2[tid];
  }
  for (int i = tid; i < TP * 64; i += 256) {
    int pi = p0 + (i >> 6); if (pi >= N) pi = N - 1;
    dAdst[i] = (float)adh[(size_t)pi * 64 + (i & 63)];
  }
  for (int e = tid; e < NEDG; e += 256) {
    int il = e / KE, k = e - il * KE;
    int gi = p0 + il; if (gi >= N) gi = N - 1;
    int s = (k == 0) ? gi : esrc[(size_t)gi * K + (k - 1)];
    if (s < 0) s = 0; if (s >= N) s = N - 1;
    srcIdx[e] = s;
#pragma unroll
    for (int d = 0; d < 3; ++d)
      dRel[e * 3 + d] = pos[(size_t)s * 3 + d] - pos[(size_t)gi * 3 + d];
  }
  __syncthreads();

  // ---- pos_nn layer1 (K=3, VALU, 4 channels/thread) -> bufA (f16, relu'd) ----
  for (int i = tid; i < NEDG * 16; i += 256) {
    int e = i >> 4, c0 = (i & 15) * 4;
    float r0 = dRel[e * 3 + 0], r1 = dRel[e * 3 + 1], r2 = dRel[e * 3 + 2];
    v4f w0 = *(const v4f*)(dPW1 + c0);
    v4f w1 = *(const v4f*)(dPW1 + 64 + c0);
    v4f w2 = *(const v4f*)(dPW1 + 128 + c0);
    v4f bb = *(const v4f*)(dPb1 + c0);
    v4h o;
#pragma unroll
    for (int j = 0; j < 4; ++j) {
      float h = bb[j] + r0 * w0[j] + r1 * w1[j] + r2 * w2[j];
      o[j] = (half_t)(h > 0.0f ? h : 0.0f);
    }
    *(v4h*)(bufA + (size_t)e * AST + c0) = o;
  }
  __syncthreads();

  // ---- delta = hidden1 @ pos_W2 + pos_b2 (WMMA) -> bufD (f32) ----
  for (int t = wave; t < NTIL; t += 8) {
    v8f acc[4];
    gemm_tile_64(bufA, AST, t * 16, wB, dPb2, lane, acc);
    store_tile_f32_lds(bufD, 64, t * 16, acc, lane);
  }
  __syncthreads();

  // ---- e = a_dst - a_src[src] + delta (4 channels/thread) -> bufA (f16) ----
  for (int i = tid; i < NEDG * 16; i += 256) {
    int e = i >> 4, c0 = (i & 15) * 4, il = e / KE;
    v4h as = *(const v4h*)(ash + (size_t)srcIdx[e] * 64 + c0);
    v4f dd = *(const v4f*)(bufD + (size_t)e * 64 + c0);
    v4f ad = *(const v4f*)(dAdst + (size_t)il * 64 + c0);
    v4h o;
#pragma unroll
    for (int j = 0; j < 4; ++j) o[j] = (half_t)(ad[j] - (float)as[j] + dd[j]);
    *(v4h*)(bufA + (size_t)e * AST + c0) = o;
  }
  __syncthreads();

  // ---- hidden2 = relu(e @ attn_W1 + b1) (WMMA, in-place per-tile) ----
  for (int t = wave; t < NTIL; t += 8) {
    v8f acc[4];
    gemm_tile_64(bufA, AST, t * 16, wB + WSW_MAT, dAb1, lane, acc);
    store_tile_f16_lds(bufA, AST, t * 16, acc, lane, true);
  }
  __syncthreads();

  // ---- logits = hidden2 @ attn_W2 + b2 (WMMA) -> bufL (f32) ----
  for (int t = wave; t < NTIL; t += 8) {
    v8f acc[4];
    gemm_tile_64(bufA, AST, t * 16, wB + 2 * WSW_MAT, dAb2, lane, acc);
    store_tile_f32_lds(bufL, 64, t * 16, acc, lane);
  }
  __syncthreads();

  // ---- softmax over k + weighted sum (2 channels/thread, TP*32 == 256 threads) ----
  {
    int i = tid;                       // exactly TP*32 items
    int il = i >> 5, c0 = (i & 31) * 2;
    int base = il * KE * 64 + c0;
    float m0 = -3.4e38f, m1 = -3.4e38f;
    for (int k = 0; k < KE; ++k) {
      v2f l = *(const v2f*)(bufL + base + k * 64);
      m0 = l[0] > m0 ? l[0] : m0;
      m1 = l[1] > m1 ? l[1] : m1;
    }
    float s0 = 0.0f, s1 = 0.0f;
    for (int k = 0; k < KE; ++k) {
      v2f l = *(const v2f*)(bufL + base + k * 64);
      v2f ex; ex[0] = __expf(l[0] - m0); ex[1] = __expf(l[1] - m1);
      *(v2f*)(bufL + base + k * 64) = ex;
      s0 += ex[0]; s1 += ex[1];
    }
    float inv0 = 1.0f / s0, inv1 = 1.0f / s1;
    float a0 = 0.0f, a1 = 0.0f;
    for (int k = 0; k < KE; ++k) {
      int e = il * KE + k;
      v2h vv = *(const v2h*)(vh + (size_t)srcIdx[e] * 64 + c0);
      v2f dd = *(const v2f*)(bufD + base + k * 64);
      v2f ex = *(const v2f*)(bufL + base + k * 64);
      a0 += ex[0] * ((float)vv[0] + dd[0]);
      a1 += ex[1] * ((float)vv[1] + dd[1]);
    }
    int gi = p0 + il;
    if (gi < N) {
      v2h o; o[0] = (half_t)(a0 * inv0); o[1] = (half_t)(a1 * inv1);
      *(v2h*)(tmph + (size_t)gi * 64 + c0) = o;
    }
  }
}

// ---------------- Kernel 3: out = tmp @ W_out + b_out ----------------
__global__ void out_kernel(const half_t* __restrict__ tmph, const float* __restrict__ b_out,
                           const half_t* __restrict__ w16, float* __restrict__ out, int N) {
  extern __shared__ char smem[];
  half_t* th = (half_t*)smem;          // 128*AST
  half_t* w  = th + 128 * AST;         // WSW_MAT (W_out)
  float*  bl = (float*)(w + WSW_MAT);  // 64
  const int tid = threadIdx.x, lane = tid & 31, wave = tid >> 5;
  const int rb = blockIdx.x * 128;

  for (int i = tid; i < WSW_MAT; i += 256) w[i] = w16[7 * WSW_MAT + i];
  if (tid < 64) bl[tid] = b_out[tid];
  for (int i = tid; i < 8192; i += 256) {
    int r = i >> 6, c = i & 63;
    int gr = rb + r; if (gr >= N) gr = N - 1;
    th[(size_t)r * AST + c] = tmph[(size_t)gr * 64 + c];
  }
  __syncthreads();

  v8f acc[4];
  gemm_tile_64(th, AST, wave * 16, w, bl, lane, acc);
  store_tile_f32_global(out, rb, wave * 16, N, acc, lane);
}

// ---------------- Host launch ----------------
extern "C" void kernel_launch(void* const* d_in, const int* in_sizes, int n_in,
                              void* d_out, int out_size, void* d_ws, size_t ws_size,
                              hipStream_t stream) {
  const float* x      = (const float*)d_in[0];
  const float* pos    = (const float*)d_in[1];
  const int*   esrc   = (const int*)d_in[2];
  const float* W_in   = (const float*)d_in[3];
  const float* b_in   = (const float*)d_in[4];
  const float* W_out  = (const float*)d_in[5];
  const float* b_out  = (const float*)d_in[6];
  const float* W_lin  = (const float*)d_in[7];
  const float* W_src  = (const float*)d_in[8];
  const float* W_dst  = (const float*)d_in[9];
  const float* posW1  = (const float*)d_in[10];
  const float* posb1  = (const float*)d_in[11];
  const float* posW2  = (const float*)d_in[12];
  const float* posb2  = (const float*)d_in[13];
  const float* attnW1 = (const float*)d_in[14];
  const float* attnb1 = (const float*)d_in[15];
  const float* attnW2 = (const float*)d_in[16];
  const float* attnb2 = (const float*)d_in[17];
  float* out = (float*)d_out;

  const int N = in_sizes[0] / 64;
  const int K = in_sizes[2] / N;

  // Workspace: [swizzled f16 weights 8*WSW_MAT][vh][a_src][a_dst][tmp] (f16 N x 64 each)
  const size_t wbytes = (size_t)8 * WSW_MAT * sizeof(half_t);   // 98304
  const size_t need = wbytes + 4 * (size_t)N * 64 * sizeof(half_t);
  if (ws_size < need) return;
  char* ws = (char*)d_ws;
  half_t* w16  = (half_t*)ws;
  half_t* vh   = (half_t*)(ws + wbytes);
  half_t* ash  = vh  + (size_t)N * 64;
  half_t* adh  = ash + (size_t)N * 64;
  half_t* tmph = adh + (size_t)N * 64;

  wcvt_kernel<<<8, 256, 0, stream>>>(W_in, W_lin, W_src, W_dst, posW2, attnW1, attnW2, W_out, w16);

  const size_t node_smem = (size_t)(2 * 128 * AST + 4 * WSW_MAT) * 2 + 256;   // 86272 B
  (void)hipFuncSetAttribute((const void*)node_kernel,
                            hipFuncAttributeMaxDynamicSharedMemorySize, (int)node_smem);
  node_kernel<<<(N + 127) / 128, 256, node_smem, stream>>>(x, b_in, w16, vh, ash, adh, N);

  const size_t edge_smem = (size_t)(2 * NROW * 64 + TP * 64 + NEDG * 3 + 192 + 4 * 64) * 4
                         + (size_t)NEDG * 4
                         + (size_t)(NROW * AST + 3 * WSW_MAT) * 2;            // 137344 B
  (void)hipFuncSetAttribute((const void*)edge_kernel,
                            hipFuncAttributeMaxDynamicSharedMemorySize, (int)edge_smem);
  edge_kernel<<<(N + TP - 1) / TP, 256, edge_smem, stream>>>(pos, esrc, ash, adh, vh,
                                                             posW1, posb1, posb2, attnb1, attnb2,
                                                             w16, tmph, N, K);

  const size_t out_smem = (size_t)(128 * AST + WSW_MAT) * 2 + 256;            // 30976 B
  (void)hipFuncSetAttribute((const void*)out_kernel,
                            hipFuncAttributeMaxDynamicSharedMemorySize, (int)out_smem);
  out_kernel<<<(N + 127) / 128, 256, out_smem, stream>>>(tmph, b_out, w16, out, N);
}
